// SwinAttntion_33809982554258
// MI455X (gfx1250) — matrix operand order repaired and checked by reading, not compiled, not run on hardware
//
#include <hip/hip_runtime.h>
#include <hip/hip_bf16.h>

// ---------------- problem constants ----------------
#define DIMC   512
#define NHEAD  16
#define HDIM   32
#define NTOK   64
#define NWIN   4096
#define NWMASK 64
#define QSCALE 0.17677669529663687f   // 32^-0.5

typedef __bf16 bf16_t;
typedef __attribute__((ext_vector_type(16))) __bf16 v16bf;
typedef __attribute__((ext_vector_type(8)))  __bf16 v8bf;
typedef __attribute__((ext_vector_type(4)))  __bf16 v4bf;
typedef __attribute__((ext_vector_type(8)))  float  v8f;

// LDS strides (in bf16 halves); all chosen so every 16-byte ds_load is aligned
// and 16 lanes reading 16 consecutive rows hit distinct bank groups.
#define XB_STRIDE  520   // 1040 B/row (row delta 260 dw -> 4 mod 64)
#define HO_STRIDE  520
#define QKV_STRIDE 40    // 80 B/row  (20 dw -> gcd(20,64)=4, conflict-free)
#define P_STRIDE   72    // 144 B/row (36 dw)
#define VT_STRIDE  72    // transposed V: [32][72], 144 B/row, 16B-aligned rows

static __device__ __forceinline__ v8f wmma_bf16(v16bf a, v16bf b, v8f c) {
  return __builtin_amdgcn_wmma_f32_16x16x32_bf16(false, a, false, b,
                                                 (short)0, c, false, false);
}

// A-matrix 16x32 bf16 fragment (contiguous k order, same as the B frags): lanes 0-15 -> M=row0+lane,
// halves 0-15 = K lo..lo+15; lanes 0-15: lo=0, lanes 16-31: lo=16.
static __device__ __forceinline__ v16bf frag_rows(const bf16_t* base, int stride,
                                                  int row0, int k0, int lane) {
  int m  = row0 + (lane & 15);
  int lo = (lane & 16) ? 16 : 0;
  const bf16_t* p = base + m * stride + k0 + lo;
  v8bf a = *(const v8bf*)p;
  v8bf b = *(const v8bf*)(p + 8);
  v16bf r;
#pragma unroll
  for (int i = 0; i < 8; ++i) { r[i] = a[i]; r[i + 8] = b[i]; }
  return r;
}

// B-matrix 32x16 bf16 fragment from a row-major [N][K] LDS buffer (B = src^T):
// lane -> N = n0+lane%16; lanes 0-15 hold K k0..k0+15, lanes 16-31 K k0+16..+31.
static __device__ __forceinline__ v16bf fragB_rows(const bf16_t* base, int stride,
                                                   int n0, int k0, int lane) {
  int n  = n0 + (lane & 15);
  int kb = k0 + ((lane & 16) ? 16 : 0);
  const bf16_t* p = base + n * stride + kb;
  v8bf a = *(const v8bf*)p;
  v8bf b = *(const v8bf*)(p + 8);
  v16bf r;
#pragma unroll
  for (int i = 0; i < 8; ++i) { r[i] = a[i]; r[i + 8] = b[i]; }
  return r;
}

// B fragment from pre-converted bf16 weights, row-major [rows][DIMC]:
// B(K,N) = W[row0+N][K]; 16 contiguous halves per lane = 2 global_load_b128.
static __device__ __forceinline__ v16bf fragB_wbf16(const bf16_t* __restrict__ w,
                                                    int row0, int k0, int lane) {
  int n  = row0 + (lane & 15);
  int kb = k0 + ((lane & 16) ? 16 : 0);
  const bf16_t* p = w + (size_t)n * DIMC + kb;
  v8bf a = *(const v8bf*)p;
  v8bf b = *(const v8bf*)(p + 8);
  v16bf r;
#pragma unroll
  for (int i = 0; i < 8; ++i) { r[i] = a[i]; r[i + 8] = b[i]; }
  return r;
}

// Store one f32 16x16 C tile (ISA C layout) to LDS as bf16: (c+bias)*scale.
// TRANSPOSE=true writes dst[col][row] instead of dst[row][col].
template <bool TRANSPOSE>
static __device__ __forceinline__ void store_tile_bf16(bf16_t* base, int stride,
                                                       int row0, int col0,
                                                       const v8f& c, float bias,
                                                       float scale, int lane) {
  int col = col0 + (lane & 15);
  int rb  = row0 + ((lane & 16) ? 8 : 0);
#pragma unroll
  for (int r = 0; r < 8; ++r) {
    bf16_t v = (bf16_t)((c[r] + bias) * scale);
    if (TRANSPOSE) base[col * stride + rb + r] = v;
    else           base[(rb + r) * stride + col] = v;
  }
}

// [64x32] = xb[64x512] @ Wbf16[rowbase:rowbase+32][512]^T + b, scaled -> LDS bf16.
// TRANSPOSE=true stores the result transposed ([32][stride]).
template <bool TRANSPOSE>
static __device__ __forceinline__ void gemm_x_w(const bf16_t* xb,
                                                const bf16_t* __restrict__ w,
                                                const float* __restrict__ bvec,
                                                int rowbase, bf16_t* dst, int dstride,
                                                float scale, int lane) {
  v8f acc[4][2] = {};
  for (int k0 = 0; k0 < DIMC; k0 += 32) {
    v16bf b0 = fragB_wbf16(w, rowbase,      k0, lane);
    v16bf b1 = fragB_wbf16(w, rowbase + 16, k0, lane);
#pragma unroll
    for (int mt = 0; mt < 4; ++mt) {
      v16bf a = frag_rows(xb, XB_STRIDE, mt * 16, k0, lane);
      acc[mt][0] = wmma_bf16(a, b0, acc[mt][0]);
      acc[mt][1] = wmma_bf16(a, b1, acc[mt][1]);
    }
  }
#pragma unroll
  for (int nt = 0; nt < 2; ++nt) {
    float bq = bvec[rowbase + nt * 16 + (lane & 15)];
#pragma unroll
    for (int mt = 0; mt < 4; ++mt)
      store_tile_bf16<TRANSPOSE>(dst, dstride, mt * 16, nt * 16, acc[mt][nt],
                                 bq, scale, lane);
  }
}

// -------- prep kernel 1: f32 -> bf16 weight conversion (vectorized x4) --------
__global__ void cvt_f32_to_bf16(const float* __restrict__ in,
                                bf16_t* __restrict__ out, int n4) {
  int i = blockIdx.x * 256 + threadIdx.x;
  if (i < n4) {
    float4 f = ((const float4*)in)[i];
    v4bf o;
    o[0] = (bf16_t)f.x; o[1] = (bf16_t)f.y; o[2] = (bf16_t)f.z; o[3] = (bf16_t)f.w;
    *(v4bf*)(out + 4 * (size_t)i) = o;
  }
}

// -------- prep kernel 2: materialize rel-pos bias rpb[h][n][m] (f32) --------
__global__ void build_rpb(const float* __restrict__ bias_table,
                          const int* __restrict__ rel_index,
                          float* __restrict__ rpb) {
  int idx = blockIdx.x * 256 + threadIdx.x;      // 16 * 4096 total
  int h  = idx >> 12;
  int nm = idx & 4095;
  rpb[idx] = bias_table[rel_index[nm] * NHEAD + h];
}

__global__ __launch_bounds__(256, 1)
void swin_attn_fused(const float* __restrict__ x,
                     const bf16_t* __restrict__ wqkv_bf,
                     const float* __restrict__ b_qkv,
                     const bf16_t* __restrict__ wproj_bf,
                     const float* __restrict__ b_proj,
                     const float* __restrict__ rpb,
                     const float* __restrict__ mask,
                     float* __restrict__ out) {
  // LDS: 66,560 + 66,560 + 122,880 = 256,000 B (< 320 KB / WGP)
  __shared__ __align__(16) bf16_t xb[NTOK * XB_STRIDE];
  __shared__ __align__(16) bf16_t ho[NTOK * HO_STRIDE];
  __shared__ __align__(16) bf16_t wsch[8][3 * NTOK * QKV_STRIDE];

  const int tid  = threadIdx.x;
  const int lane = tid & 31;
  const int wave = tid >> 5;
  const int win  = blockIdx.x;

  // ---- Phase 0: stage x window into LDS as bf16 ----
  const float4* x4 = (const float4*)(x + (size_t)win * NTOK * DIMC);
  for (int i = tid; i < NTOK * (DIMC / 4); i += 256) {
    int m = i >> 7;               // / 128
    int c = (i & 127) << 2;
    float4 f = x4[i];
    v4bf o;
    o[0] = (bf16_t)f.x; o[1] = (bf16_t)f.y; o[2] = (bf16_t)f.z; o[3] = (bf16_t)f.w;
    *(v4bf*)&xb[m * XB_STRIDE + c] = o;
  }
  // warm L2 with the projection weights while attention runs (global_prefetch_b8)
  for (int i = tid; i < DIMC; i += 256)
    __builtin_prefetch(wproj_bf + (size_t)i * DIMC, 0, 1);
  __syncthreads();

  bf16_t* lds_q  = &wsch[wave][0];
  bf16_t* lds_k  = lds_q + NTOK * QKV_STRIDE;
  bf16_t* lds_vT = lds_k + NTOK * QKV_STRIDE;  // [HDIM][VT_STRIDE], 4608 B
  bf16_t* lds_p  = lds_q;  // probs overlay q+k once their fragments are in VGPRs

  const float* maskw = mask + (size_t)(win & (NWMASK - 1)) * NTOK * NTOK;
  const int colm   = lane & 15;
  const int rowoff = (lane & 16) ? 8 : 0;

  for (int hi = 0; hi < 2; ++hi) {
    const int h = wave + (hi << 3);
    const float* rpbh = rpb + (size_t)h * NTOK * NTOK;

    // ---- QKV projection for this head (q pre-scaled, v stored transposed) ----
    gemm_x_w<false>(xb, wqkv_bf, b_qkv,        h * HDIM, lds_q,  QKV_STRIDE, QSCALE, lane);
    gemm_x_w<false>(xb, wqkv_bf, b_qkv,  512 + h * HDIM, lds_k,  QKV_STRIDE, 1.0f,  lane);
    gemm_x_w<true >(xb, wqkv_bf, b_qkv, 1024 + h * HDIM, lds_vT, VT_STRIDE,  1.0f,  lane);

    // ---- scores = q @ k^T  (single K=32 WMMA step) ----
    v16bf qa[4], kf[4];
#pragma unroll
    for (int mt = 0; mt < 4; ++mt) qa[mt] = frag_rows (lds_q, QKV_STRIDE, mt * 16, 0, lane);
#pragma unroll
    for (int nt = 0; nt < 4; ++nt) kf[nt] = fragB_rows(lds_k, QKV_STRIDE, nt * 16, 0, lane);
    v8f s[4][4] = {};
#pragma unroll
    for (int mt = 0; mt < 4; ++mt)
#pragma unroll
      for (int nt = 0; nt < 4; ++nt)
        s[mt][nt] = wmma_bf16(qa[mt], kf[nt], s[mt][nt]);

    // ---- precomputed rel-pos bias + mask + row softmax ----
#pragma unroll
    for (int mt = 0; mt < 4; ++mt) {
#pragma unroll
      for (int r = 0; r < 8; ++r) {
        const int n = mt * 16 + rowoff + r;
        float vals[4];
#pragma unroll
        for (int nt = 0; nt < 4; ++nt) {
          const int nm = n * NTOK + nt * 16 + colm;
          vals[nt] = s[mt][nt][r] + rpbh[nm] + maskw[nm];
        }
        float rm = fmaxf(fmaxf(vals[0], vals[1]), fmaxf(vals[2], vals[3]));
        rm = fmaxf(rm, __shfl_xor(rm, 1, 32));
        rm = fmaxf(rm, __shfl_xor(rm, 2, 32));
        rm = fmaxf(rm, __shfl_xor(rm, 4, 32));
        rm = fmaxf(rm, __shfl_xor(rm, 8, 32));
        float sum = 0.f;
#pragma unroll
        for (int nt = 0; nt < 4; ++nt) { vals[nt] = __expf(vals[nt] - rm); sum += vals[nt]; }
        sum += __shfl_xor(sum, 1, 32);
        sum += __shfl_xor(sum, 2, 32);
        sum += __shfl_xor(sum, 4, 32);
        sum += __shfl_xor(sum, 8, 32);
        const float inv = 1.0f / sum;
#pragma unroll
        for (int nt = 0; nt < 4; ++nt) s[mt][nt][r] = vals[nt] * inv;
      }
    }
    // stage probs (bf16) for re-load in A-fragment layout
#pragma unroll
    for (int mt = 0; mt < 4; ++mt)
#pragma unroll
      for (int nt = 0; nt < 4; ++nt)
        store_tile_bf16<false>(lds_p, P_STRIDE, mt * 16, nt * 16, s[mt][nt],
                               0.f, 1.f, lane);

    // ---- head_out = P[64x64] @ V[64x32]  (V^T in LDS -> vector B-frag loads) ----
    v8f o[4][2] = {};
#pragma unroll
    for (int ks = 0; ks < 2; ++ks) {
      const int k0 = ks * 32;
      v16bf vb0 = fragB_rows(lds_vT, VT_STRIDE,  0, k0, lane);
      v16bf vb1 = fragB_rows(lds_vT, VT_STRIDE, 16, k0, lane);
#pragma unroll
      for (int mt = 0; mt < 4; ++mt) {
        v16bf pa = frag_rows(lds_p, P_STRIDE, mt * 16, k0, lane);
        o[mt][0] = wmma_bf16(pa, vb0, o[mt][0]);
        o[mt][1] = wmma_bf16(pa, vb1, o[mt][1]);
      }
    }
#pragma unroll
    for (int mt = 0; mt < 4; ++mt)
#pragma unroll
      for (int nt = 0; nt < 2; ++nt)
        store_tile_bf16<false>(ho, HO_STRIDE, mt * 16, h * HDIM + nt * 16,
                               o[mt][nt], 0.f, 1.f, lane);
  }

  __syncthreads();

  // ---- output projection: each wave owns 64 output columns ----
  const int ncol0 = wave * 64;
  v8f acc[4][4] = {};
  for (int k0 = 0; k0 < DIMC; k0 += 32) {
    v16bf bf[4];
#pragma unroll
    for (int nt = 0; nt < 4; ++nt)
      bf[nt] = fragB_wbf16(wproj_bf, ncol0 + nt * 16, k0, lane);
#pragma unroll
    for (int mt = 0; mt < 4; ++mt) {
      v16bf a = frag_rows(ho, HO_STRIDE, mt * 16, k0, lane);
#pragma unroll
      for (int nt = 0; nt < 4; ++nt)
        acc[mt][nt] = wmma_bf16(a, bf[nt], acc[mt][nt]);
    }
  }
  float* outw = out + (size_t)win * NTOK * DIMC;
#pragma unroll
  for (int nt = 0; nt < 4; ++nt) {
    const int col = ncol0 + nt * 16 + colm;
    const float bp = b_proj[col];
#pragma unroll
    for (int mt = 0; mt < 4; ++mt) {
      const int rb = mt * 16 + rowoff;
#pragma unroll
      for (int r = 0; r < 8; ++r)
        outw[(size_t)(rb + r) * DIMC + col] = acc[mt][nt][r] + bp;
    }
  }
}

extern "C" void kernel_launch(void* const* d_in, const int* in_sizes, int n_in,
                              void* d_out, int out_size, void* d_ws, size_t ws_size,
                              hipStream_t stream) {
  const float* x          = (const float*)d_in[0];
  const float* w_qkv      = (const float*)d_in[1];
  const float* b_qkv      = (const float*)d_in[2];
  const float* w_proj     = (const float*)d_in[3];
  const float* b_proj     = (const float*)d_in[4];
  const float* bias_table = (const float*)d_in[5];
  const float* mask       = (const float*)d_in[6];
  const int*   rel_index  = (const int*)d_in[7];
  float* out = (float*)d_out;

  // workspace layout (all 16B aligned):
  //   [0, 1.5MB)      w_qkv  as bf16 [1536][512]
  //   [1.5MB, 2MB)    w_proj as bf16 [512][512]
  //   [2MB, 2.25MB)   rpb f32 [16][64][64]
  char* ws = (char*)d_ws;
  bf16_t* wqkv_bf  = (bf16_t*)ws;
  bf16_t* wproj_bf = (bf16_t*)(ws + (size_t)3 * DIMC * DIMC * sizeof(bf16_t));
  float*  rpb      = (float*) (ws + (size_t)4 * DIMC * DIMC * sizeof(bf16_t));

  // prep: convert weights to bf16, materialize rel-pos bias per head
  {
    int n4 = (3 * DIMC * DIMC) / 4;           // w_qkv: 786,432 floats
    hipLaunchKernelGGL(cvt_f32_to_bf16, dim3((n4 + 255) / 256), dim3(256), 0, stream,
                       w_qkv, wqkv_bf, n4);
  }
  {
    int n4 = (DIMC * DIMC) / 4;               // w_proj: 262,144 floats
    hipLaunchKernelGGL(cvt_f32_to_bf16, dim3((n4 + 255) / 256), dim3(256), 0, stream,
                       w_proj, wproj_bf, n4);
  }
  hipLaunchKernelGGL(build_rpb, dim3((NHEAD * NTOK * NTOK) / 256), dim3(256), 0, stream,
                     bias_table, rel_index, rpb);

  hipLaunchKernelGGL(swin_attn_fused, dim3(NWIN), dim3(256), 0, stream,
                     x, wqkv_bf, b_qkv, wproj_bf, b_proj, rpb, mask, out);
}
